// DecoderLSTM_29695403885094
// MI455X (gfx1250) — compile-verified, matrix operand order checked by reading, not run on hardware
//
#include <hip/hip_runtime.h>
#include <hip/hip_bf16.h>
#include <math.h>

typedef __attribute__((ext_vector_type(16))) __bf16 v16bf;
typedef __attribute__((ext_vector_type(8)))  float  v8f;

#define AST 36   // LDS row stride (floats) for A tiles: 144B -> 16B-aligned frags, conflict-free
#define BST 36   // same for B tiles

__device__ __forceinline__ float sigmoidf_(float x) {
    return 1.0f / (1.0f + __expf(-x));
}

// Async global->LDS copy of 16B per lane, tracked on ASYNCcnt (CDNA5 path).
// VDST VGPR carries the per-lane LDS byte offset (low 32 bits of the generic
// shared-space address per the LDS aperture mapping).
__device__ __forceinline__ void async_load_b128(const void* gaddr, void* lds_dst) {
    uint32_t lds_off = (uint32_t)(uintptr_t)lds_dst;
    asm volatile("global_load_async_to_lds_b128 %0, %1, off"
                 :: "v"(lds_off), "v"(gaddr) : "memory");
}

// Build a bf16 WMMA fragment from 16 consecutive f32 in LDS (use-time truncation;
// clang fuses the pairs into v_cvt_pk_bf16_f32, which co-executes with WMMA).
__device__ __forceinline__ v16bf load_frag_bf16(const float* src) {
    float4 x0 = *reinterpret_cast<const float4*>(src);
    float4 x1 = *reinterpret_cast<const float4*>(src + 4);
    float4 x2 = *reinterpret_cast<const float4*>(src + 8);
    float4 x3 = *reinterpret_cast<const float4*>(src + 12);
    v16bf f;
    f[0]  = (__bf16)x0.x; f[1]  = (__bf16)x0.y; f[2]  = (__bf16)x0.z; f[3]  = (__bf16)x0.w;
    f[4]  = (__bf16)x1.x; f[5]  = (__bf16)x1.y; f[6]  = (__bf16)x1.z; f[7]  = (__bf16)x1.w;
    f[8]  = (__bf16)x2.x; f[9]  = (__bf16)x2.y; f[10] = (__bf16)x2.z; f[11] = (__bf16)x2.w;
    f[12] = (__bf16)x3.x; f[13] = (__bf16)x3.y; f[14] = (__bf16)x3.z; f[15] = (__bf16)x3.w;
    return f;
}

// D(MxN) = A0(MxK0) @ B0(NxK0)^T [+ A1(MxK1) @ B1(NxK1)^T] + bias0 [+ bias1]
// Block tile 128(M) x 64(N), K-step 32, 256 threads = 8 waves (4x2 grid), each
// wave a 32x32 sub-tile via 2x2 v_wmma_f32_16x16x32_bf16. Tiles staged with
// double-buffered GLOBAL_LOAD_ASYNC_TO_LDS_B128 (6 per wave per tile) and
// s_wait_asynccnt pipelining. Requires M%128==0, N%64==0, K%32==0.
__global__ __launch_bounds__(256)
void gemm_nt_wmma(const float* __restrict__ A0, const float* __restrict__ B0, int K0,
                  const float* __restrict__ A1, const float* __restrict__ B1, int K1,
                  const float* __restrict__ bias0, const float* __restrict__ bias1,
                  float* __restrict__ D, int M, int N)
{
    __shared__ __align__(16) float Abuf[2][128 * AST];
    __shared__ __align__(16) float Bbuf[2][ 64 * BST];

    const int tid  = threadIdx.x;
    const int wid  = tid >> 5;
    const int lane = tid & 31;
    const int lr   = lane & 15;   // row/col within 16
    const int kh   = lane >> 4;   // K-half of the fragment

    const int tileN = blockIdx.x * 64;
    const int tileM = blockIdx.y * 128;
    const int mBase = (wid >> 1) * 32;
    const int nBase = (wid &  1) * 32;

    v8f acc[2][2] = {};

    const int nt0    = K0 >> 5;
    const int nt1    = A1 ? (K1 >> 5) : 0;
    const int ntiles = nt0 + nt1;

    auto stage = [&](int ti, int buf) {
        const float* A; const float* B; int K, k0;
        if (ti < nt0) { A = A0; B = B0; K = K0; k0 = ti << 5; }
        else          { A = A1; B = B1; K = K1; k0 = (ti - nt0) << 5; }
        #pragma unroll
        for (int i = 0; i < 4; ++i) {           // A tile: 128x32 f32 = 4 issues/thread
            int f4 = tid + (i << 8), row = f4 >> 3, c4 = f4 & 7;
            async_load_b128(A + (size_t)(tileM + row) * K + k0 + c4 * 4,
                            &Abuf[buf][row * AST + c4 * 4]);
        }
        #pragma unroll
        for (int i = 0; i < 2; ++i) {           // B tile: 64x32 f32 = 2 issues/thread
            int f4 = tid + (i << 8), row = f4 >> 3, c4 = f4 & 7;
            async_load_b128(B + (size_t)(tileN + row) * K + k0 + c4 * 4,
                            &Bbuf[buf][row * BST + c4 * 4]);
        }
    };

    stage(0, 0);
    for (int ti = 0; ti < ntiles; ++ti) {
        const int cur = ti & 1;
        if (ti + 1 < ntiles) {
            stage(ti + 1, cur ^ 1);             // prefetch next tile while computing
            asm volatile("s_wait_asynccnt 0x6" ::: "memory");   // prev 6 done (in-order)
        } else {
            asm volatile("s_wait_asynccnt 0x0" ::: "memory");
        }
        __syncthreads();                        // all waves' tile-ti LDS writes visible

        v16bf afrag[2], bfrag[2];
        #pragma unroll
        for (int mt = 0; mt < 2; ++mt)
            afrag[mt] = load_frag_bf16(&Abuf[cur][(mBase + mt * 16 + lr) * AST + kh * 16]);
        #pragma unroll
        for (int nt = 0; nt < 2; ++nt)
            bfrag[nt] = load_frag_bf16(&Bbuf[cur][(nBase + nt * 16 + lr) * BST + kh * 16]);

        #pragma unroll
        for (int mt = 0; mt < 2; ++mt)
            #pragma unroll
            for (int nt = 0; nt < 2; ++nt)
                acc[mt][nt] = __builtin_amdgcn_wmma_f32_16x16x32_bf16(
                    false, afrag[mt], false, bfrag[nt],
                    (short)0, acc[mt][nt], false, false);

        __syncthreads();                        // protect buf[cur] before it is re-staged
    }

    // C/D layout: lane (lr) = N col; VGPR r = row r (lanes 0-15) or 8+r (lanes 16-31)
    #pragma unroll
    for (int nt = 0; nt < 2; ++nt) {
        int col = tileN + nBase + nt * 16 + lr;
        float bv = bias0 ? bias0[col] : 0.0f;
        if (bias1) bv += bias1[col];
        #pragma unroll
        for (int mt = 0; mt < 2; ++mt) {
            #pragma unroll
            for (int r = 0; r < 8; ++r) {
                int row = tileM + mBase + mt * 16 + kh * 8 + r;
                D[(size_t)row * N + col] = acc[mt][nt][r] + bv;
            }
        }
    }
}

// c = sig(f)*c0 + sig(i)*tanh(g); h = sig(o)*tanh(c). Also scatter h into cat[:, :H].
__global__ __launch_bounds__(256)
void lstm_pointwise(const float* __restrict__ gates, const float* __restrict__ c0,
                    float* __restrict__ h_out, float* __restrict__ c_out,
                    float* __restrict__ cat, int H)
{
    int idx = blockIdx.x * 256 + threadIdx.x;   // B*H threads, H == 1024
    int b   = idx >> 10;
    int hI  = idx & 1023;
    const float* g = gates + (size_t)b * 4 * H;
    float ig = sigmoidf_(g[hI]);
    float fg = sigmoidf_(g[H + hI]);
    float gg = tanhf(g[2 * H + hI]);
    float og = sigmoidf_(g[3 * H + hI]);
    float c  = fg * c0[idx] + ig * gg;
    float hh = og * tanhf(c);
    h_out[idx] = hh;
    c_out[idx] = c;
    cat[(size_t)b * 2 * H + hI] = hh;
}

// One block per batch row. Single pass over enc[b]: each enc row is read once and
// reused (registers) for both the score dot-product and the online-softmax-weighted
// context accumulation. Masked scores are set to 1e-10 (matching the reference,
// which softmaxes over all T including masked slots).
__global__ __launch_bounds__(256)
void attn_softmax_ctx(const float* __restrict__ h, const float* __restrict__ enc,
                      const int* __restrict__ xlens, float* __restrict__ attn,
                      float* __restrict__ cat, int T, int H)
{
    const int b    = blockIdx.x;
    const int tid  = threadIdx.x;
    const int wid  = tid >> 5;
    const int lane = tid & 31;

    __shared__ float s_sc[128];   // raw scores (T <= 128)
    __shared__ float red[8];      // per-wave partials

    float hreg[4], acc[4] = {0.f, 0.f, 0.f, 0.f};
    #pragma unroll
    for (int j = 0; j < 4; ++j) hreg[j] = h[(size_t)b * H + tid + j * 256];

    const int xlen = xlens[b];
    float m = -INFINITY, l = 0.0f;

    for (int t = 0; t < T; ++t) {
        const float* er = enc + ((size_t)b * T + t) * H;
        // prefetch row t+2 (256 threads x 16B stride covers the whole 4KB row)
        if (t + 2 < T) __builtin_prefetch(er + 2 * H + tid * 4, 0, 1);

        float e[4];
        float p = 0.0f;
        #pragma unroll
        for (int j = 0; j < 4; ++j) { e[j] = er[tid + j * 256]; p += e[j] * hreg[j]; }
        // wave32 reduction
        #pragma unroll
        for (int off = 16; off > 0; off >>= 1) p += __shfl_xor(p, off, 32);
        if (lane == 0) red[wid] = p;
        __syncthreads();
        if (tid == 0) {
            float s = red[0];
            #pragma unroll
            for (int w = 1; w < 8; ++w) s += red[w];
            if (t >= xlen) s = 1e-10f;
            s_sc[t] = s;
        }
        __syncthreads();
        float s     = s_sc[t];
        float mnew  = fmaxf(m, s);
        float scale = __expf(m - mnew);    // 0 on first iteration (m = -inf)
        float pt    = __expf(s - mnew);
        l = l * scale + pt;
        m = mnew;
        #pragma unroll
        for (int j = 0; j < 4; ++j) acc[j] = acc[j] * scale + pt * e[j];
    }

    float inv_l = 1.0f / l;   // m, l replicated identically across all threads
    #pragma unroll
    for (int j = 0; j < 4; ++j)
        cat[(size_t)b * (2 * H) + H + tid + j * 256] = acc[j] * inv_l;
    if (tid < T)
        attn[(size_t)b * T + tid] = __expf(s_sc[tid] - m) * inv_l;
}

extern "C" void kernel_launch(void* const* d_in, const int* in_sizes, int n_in,
                              void* d_out, int out_size, void* d_ws, size_t ws_size,
                              hipStream_t stream) {
    const float* inputs = (const float*)d_in[0];
    const float* h0     = (const float*)d_in[1];
    const float* c0     = (const float*)d_in[2];
    const float* enc    = (const float*)d_in[3];
    const int*   xlens  = (const int*)  d_in[4];
    // d_in[5] = batch_size scalar (unused; shapes are fixed)
    const float* W_ih   = (const float*)d_in[6];
    const float* W_hh   = (const float*)d_in[7];
    const float* b_ih   = (const float*)d_in[8];
    const float* b_hh   = (const float*)d_in[9];
    const float* Wc     = (const float*)d_in[10];
    const float* bc     = (const float*)d_in[11];

    const int B = 1024, T = 90, H = 1024, OUT = 512;

    // d_out layout (return-order flatten): output | h | c | attn_weights
    float* out   = (float*)d_out;
    float* h_out = out   + (size_t)B * OUT;
    float* c_out = h_out + (size_t)B * H;
    float* attn  = c_out + (size_t)B * H;

    // workspace: gates (B x 4H) then cat (B x 2H)
    float* gates = (float*)d_ws;
    float* cat   = gates + (size_t)B * 4 * H;

    // 1) gates = inputs @ W_ih^T + h0 @ W_hh^T + b_ih + b_hh   (M=1024, N=4096)
    gemm_nt_wmma<<<dim3((4 * H) / 64, B / 128), 256, 0, stream>>>(
        inputs, W_ih, OUT, h0, W_hh, H, b_ih, b_hh, gates, B, 4 * H);

    // 2) LSTM pointwise -> h, c (into d_out) and cat[:, :H] = h
    lstm_pointwise<<<(B * H) / 256, 256, 0, stream>>>(gates, c0, h_out, c_out, cat, H);

    // 3) fused scores + masked online-softmax + context -> attn, cat[:, H:]
    attn_softmax_ctx<<<B, 256, 0, stream>>>(h_out, enc, xlens, attn, cat, T, H);

    // 4) output = cat @ Wc^T + bc   (M=1024, N=512, K=2048)
    gemm_nt_wmma<<<dim3(OUT / 64, B / 128), 256, 0, stream>>>(
        cat, Wc, 2 * H, nullptr, nullptr, 0, bc, nullptr, out, B, OUT);
}
